// SAE_17755394801989
// MI455X (gfx1250) — matrix operand order repaired and checked by reading, not compiled, not run on hardware
//
#include <hip/hip_runtime.h>
#include <hip/hip_bf16.h>

// ---------------------------------------------------------------------------
// Problem constants (fixed by the reference)
// ---------------------------------------------------------------------------
#define D_IN    2048
#define D_HID   16384
#define NTOK    8192
#define TOPK    64

typedef __attribute__((ext_vector_type(16))) __bf16 v16bf;
typedef __attribute__((ext_vector_type(8)))  __bf16 v8bf;
typedef __attribute__((ext_vector_type(8)))  float  v8f;
typedef __attribute__((ext_vector_type(4)))  int    v4i;

#define AS1 __attribute__((address_space(1)))
#define AS3 __attribute__((address_space(3)))

// Async direct-to-LDS staging (gfx1250 path), with sync fallback.
#if defined(__has_builtin)
#  if __has_builtin(__builtin_amdgcn_global_load_async_to_lds_b128) && \
      __has_builtin(__builtin_amdgcn_s_wait_asynccnt)
#    define USE_ASYNC_LDS 1
#  endif
#endif
#ifndef USE_ASYNC_LDS
#  define USE_ASYNC_LDS 0
#endif

// ---------------------------------------------------------------------------
// Workspace layout (bytes)
// ---------------------------------------------------------------------------
#define OFF_XB   ((size_t)0)                         // bf16 x-bias   [8192 x 2048]   33.5 MB
#define OFF_WB   ((size_t)33554432)                  // bf16 W_enc    [16384 x 2048]  67.1 MB
#define OFF_PRE  ((size_t)100663296)                 // f32 pre       [8192 x 16384] 536.9 MB
#define OFF_WDT  ((size_t)637534208)                 // f32 W_dec^T   [16384 x 2048] 134.2 MB
#define OFF_CV   ((size_t)771751936)                 // f32 compact vals [8192 x 64]   2.1 MB
#define OFF_CI   ((size_t)773849088)                 // i32 compact idx  [8192 x 64]   2.1 MB

// ---------------------------------------------------------------------------
// 1) Fold bias into x and convert to bf16 (8 elems / thread, 16B stores)
// ---------------------------------------------------------------------------
__global__ __launch_bounds__(256) void prep_x_kernel(const float* __restrict__ x,
                                                     const float* __restrict__ bias,
                                                     __bf16* __restrict__ xb) {
  const size_t n8 = (size_t)NTOK * D_IN / 8;
  for (size_t i = (size_t)blockIdx.x * 256 + threadIdx.x; i < n8; i += (size_t)gridDim.x * 256) {
    const size_t base = i * 8;
    const float4 a = ((const float4*)x)[2 * i + 0];
    const float4 b = ((const float4*)x)[2 * i + 1];
    const int bo = (int)(base & (D_IN - 1));   // D_IN divides 8-groups evenly
    v8bf o;
    o[0] = (__bf16)(a.x - bias[bo + 0]);
    o[1] = (__bf16)(a.y - bias[bo + 1]);
    o[2] = (__bf16)(a.z - bias[bo + 2]);
    o[3] = (__bf16)(a.w - bias[bo + 3]);
    o[4] = (__bf16)(b.x - bias[bo + 4]);
    o[5] = (__bf16)(b.y - bias[bo + 5]);
    o[6] = (__bf16)(b.z - bias[bo + 6]);
    o[7] = (__bf16)(b.w - bias[bo + 7]);
    *(v8bf*)(xb + base) = o;
  }
}

// 2) Convert W_enc to bf16
__global__ __launch_bounds__(256) void prep_w_kernel(const float* __restrict__ W,
                                                     __bf16* __restrict__ wb) {
  const size_t n8 = (size_t)D_HID * D_IN / 8;
  for (size_t i = (size_t)blockIdx.x * 256 + threadIdx.x; i < n8; i += (size_t)gridDim.x * 256) {
    const float4 a = ((const float4*)W)[2 * i + 0];
    const float4 b = ((const float4*)W)[2 * i + 1];
    v8bf o;
    o[0] = (__bf16)a.x; o[1] = (__bf16)a.y; o[2] = (__bf16)a.z; o[3] = (__bf16)a.w;
    o[4] = (__bf16)b.x; o[5] = (__bf16)b.y; o[6] = (__bf16)b.z; o[7] = (__bf16)b.w;
    *(v8bf*)(wb + i * 8) = o;
  }
}

// ---------------------------------------------------------------------------
// 3) Transpose W_dec [D_IN x D_HID] -> WdT [D_HID x D_IN] for coalesced
//    row gathers in the sparse decoder.
// ---------------------------------------------------------------------------
__global__ __launch_bounds__(256) void transpose_wdec_kernel(const float* __restrict__ W,
                                                             float* __restrict__ WT) {
  __shared__ float tile[32][33];
  const int h0 = blockIdx.x * 32;           // along D_HID
  const int d0 = blockIdx.y * 32;           // along D_IN
  const int tx = threadIdx.x & 31;
  const int ty = threadIdx.x >> 5;          // 0..7
  #pragma unroll
  for (int r = ty; r < 32; r += 8)
    tile[r][tx] = W[(size_t)(d0 + r) * D_HID + h0 + tx];
  __syncthreads();
  #pragma unroll
  for (int r = ty; r < 32; r += 8)
    WT[(size_t)(h0 + r) * D_IN + d0 + tx] = tile[tx][r];
}

// ---------------------------------------------------------------------------
// 4) Encoder GEMM: pre[M=8192, N=16384] = xb[M,K] * wb[N,K]^T + b_enc
//    bf16 WMMA 16x16x32, f32 accum. Block tile 128x128x32, 8 waves (2x4),
//    each wave 64x32 = 8 WMMAs/K-step. Double-buffered LDS fed by
//    GLOBAL_LOAD_ASYNC_TO_LDS_B128 (ASYNCcnt pipeline).
// ---------------------------------------------------------------------------
__device__ __forceinline__ v16bf combine16(v8bf lo, v8bf hi) {
  v16bf r;
  #pragma unroll
  for (int i = 0; i < 8; ++i) { r[i] = lo[i]; r[i + 8] = hi[i]; }
  return r;
}

#define LDS_ROW_B  80                         // 64B data + 16B pad (bank spread)
#define MAT_BYTES  (128 * LDS_ROW_B)          // 10240 per matrix tile
#define TILE_BYTES (2 * MAT_BYTES)            // 20480 per buffer (A + B)

__global__ __launch_bounds__(256) void enc_gemm_kernel(const __bf16* __restrict__ xb,
                                                       const __bf16* __restrict__ wb,
                                                       const float* __restrict__ b_enc,
                                                       float* __restrict__ pre) {
  __shared__ __align__(16) unsigned char smem[2 * TILE_BYTES];   // 40 KB

  const int tid    = threadIdx.x;
  const int lane   = tid & 31;
  const int wid    = tid >> 5;           // 0..7
  const int wave_m = wid >> 2;           // 0..1  -> 64-row slab
  const int wave_n = wid & 3;            // 0..3  -> 32-col slab
  const int half   = lane >> 4;          // wave32 lane half
  const int lr     = lane & 15;

  const int m0 = blockIdx.y * 128;
  const int n0 = blockIdx.x * 128;

  v8f acc[4][2];
  #pragma unroll
  for (int wm = 0; wm < 4; ++wm)
    #pragma unroll
    for (int wn = 0; wn < 2; ++wn) acc[wm][wn] = 0;

  // ---- stage one K-tile pair (A+B) into buffer `buf`; 4 async ops / wave ----
  auto stage = [&](int buf, int k0) {
    unsigned char* ldsA = smem + buf * TILE_BYTES;
    unsigned char* ldsB = ldsA + MAT_BYTES;
    #pragma unroll
    for (int t = 0; t < 2; ++t) {
      const int c  = tid + t * 256;           // 0..511
      const int r  = c >> 2;                  // row 0..127
      const int co = (c & 3) * 16;            // 16B chunk in row
      const unsigned char* ga = (const unsigned char*)xb +
                                ((size_t)(m0 + r) * D_IN + k0) * 2 + co;
      const unsigned char* gb = (const unsigned char*)wb +
                                ((size_t)(n0 + r) * D_IN + k0) * 2 + co;
      unsigned char* la = ldsA + r * LDS_ROW_B + co;
      unsigned char* lb = ldsB + r * LDS_ROW_B + co;
#if USE_ASYNC_LDS
      __builtin_amdgcn_global_load_async_to_lds_b128((AS1 v4i*)ga, (AS3 v4i*)la, 0, 0);
      __builtin_amdgcn_global_load_async_to_lds_b128((AS1 v4i*)gb, (AS3 v4i*)lb, 0, 0);
#else
      *(uint4*)la = *(const uint4*)ga;
      *(uint4*)lb = *(const uint4*)gb;
#endif
    }
  };

  // ---- consume buffer `buf`: per-lane fragment loads per the CDNA5 16-bit
  //      WMMA VGPR layouts, then 8 WMMAs ----
  auto compute = [&](int buf) {
    const unsigned char* ldsA = smem + buf * TILE_BYTES;
    const unsigned char* ldsB = ldsA + MAT_BYTES;
    v16bf afr[4], bfr[2];
    #pragma unroll
    for (int wm = 0; wm < 4; ++wm) {
      const int row = wave_m * 64 + wm * 16 + lr;           // A: M row = lane
      const unsigned char* pa = ldsA + row * LDS_ROW_B;
      // half 0: K{0..7,16..23}; half 1: K{8..15,24..31}
      v8bf alo = *(const v8bf*)(pa + half * 16);
      v8bf ahi = *(const v8bf*)(pa + 32 + half * 16);
      afr[wm] = combine16(alo, ahi);
    }
    #pragma unroll
    for (int wn = 0; wn < 2; ++wn) {
      const int nn = wave_n * 32 + wn * 16 + lr;            // B: N col = lane
      const unsigned char* pb = ldsB + nn * LDS_ROW_B + half * 16; // K{0..7,16..23} / K{8..15,24..31}
      v8bf blo = *(const v8bf*)(pb);
      v8bf bhi = *(const v8bf*)(pb + 32);
      bfr[wn] = combine16(blo, bhi);
    }
    #pragma unroll
    for (int wm = 0; wm < 4; ++wm)
      #pragma unroll
      for (int wn = 0; wn < 2; ++wn)
        acc[wm][wn] = __builtin_amdgcn_wmma_f32_16x16x32_bf16(
            false, afr[wm], false, bfr[wn], (short)0, acc[wm][wn], false, false);
  };

  // ---- double-buffered pipeline over K ----
  stage(0, 0);
  for (int k0 = 0; k0 < D_IN - 32; k0 += 32) {
    const int cur = (k0 >> 5) & 1;
    stage(cur ^ 1, k0 + 32);                 // prefetch next tile (+4 outstanding)
#if USE_ASYNC_LDS
    __builtin_amdgcn_s_wait_asynccnt(4);     // in-order: oldest 4 (cur tile) done
#endif
    __syncthreads();
    compute(cur);
    __syncthreads();
  }
#if USE_ASYNC_LDS
  __builtin_amdgcn_s_wait_asynccnt(0);       // drain final tile
#endif
  __syncthreads();
  compute(((D_IN - 32) >> 5) & 1);

  // ---- epilogue: VGPR i holds M=i (lanes 0-15) / M=i+8 (lanes 16-31) ----
  #pragma unroll
  for (int wm = 0; wm < 4; ++wm)
    #pragma unroll
    for (int wn = 0; wn < 2; ++wn) {
      const int colg = n0 + wave_n * 32 + wn * 16 + lr;
      const float be = b_enc[colg];
      #pragma unroll
      for (int i = 0; i < 8; ++i) {
        const int rowg = m0 + wave_m * 64 + wm * 16 + i + half * 8;
        pre[(size_t)rowg * D_HID + colg] = acc[wm][wn][i] + be;
      }
    }
}

// ---------------------------------------------------------------------------
// 5) Exact per-row top-64 via 4-pass radix select on monotonic float keys.
//    Writes dense f row (zeros elsewhere) + compact (val,idx) list.
// ---------------------------------------------------------------------------
__device__ __forceinline__ unsigned fkey(float v) {
  unsigned b = __float_as_uint(v);
  return (b & 0x80000000u) ? ~b : (b | 0x80000000u);
}

__global__ __launch_bounds__(256) void topk_kernel(const float* __restrict__ pre,
                                                   float* __restrict__ f,
                                                   float* __restrict__ cval,
                                                   int* __restrict__ cidx) {
  __shared__ unsigned hist[256];
  __shared__ unsigned sh_prefix, sh_remaining, sh_count;
  const int row = blockIdx.x;
  const int tid = threadIdx.x;
  const float* p = pre + (size_t)row * D_HID;

  if (tid == 0) { sh_prefix = 0u; sh_remaining = TOPK; sh_count = 0u; }

  for (int pass = 0; pass < 4; ++pass) {
    const int shift = 24 - 8 * pass;
    hist[tid] = 0u;
    __syncthreads();
    const unsigned pfx = sh_prefix;
    for (int i = tid; i < D_HID; i += 256) {
      const unsigned k = fkey(p[i]);
      if (pass == 0 || (k >> (shift + 8)) == pfx)
        atomicAdd(&hist[(k >> shift) & 0xFFu], 1u);
    }
    __syncthreads();
    if (tid == 0) {
      unsigned rem = sh_remaining, c = 0u;
      int d = 0;
      for (int b = 255; b >= 0; --b) {
        const unsigned h = hist[b];
        if (c + h >= rem) { d = b; sh_remaining = rem - c; break; }
        c += h;
      }
      sh_prefix = (sh_prefix << 8) | (unsigned)d;
    }
    __syncthreads();
  }

  const unsigned T = sh_prefix;   // key of the 64th-largest element
  float* fr = f + (size_t)row * D_HID;
  for (int i = tid; i < D_HID; i += 256) {
    const float v = p[i];
    const bool sel = (fkey(v) >= T) && (v > 0.0f);
    fr[i] = sel ? v : 0.0f;
    if (sel) {
      const unsigned s = atomicAdd(&sh_count, 1u);
      if (s < TOPK) { cval[(size_t)row * TOPK + s] = v; cidx[(size_t)row * TOPK + s] = i; }
    }
  }
  __syncthreads();
  const unsigned cnt = sh_count;
  if (tid < TOPK && tid >= cnt) {          // pad so decoder loops exactly 64
    cval[(size_t)row * TOPK + tid] = 0.0f;
    cidx[(size_t)row * TOPK + tid] = 0;
  }
}

// ---------------------------------------------------------------------------
// 6) Sparse decode: x_hat[n,:] = bias + sum_j val_j * WdT[idx_j, :]
// ---------------------------------------------------------------------------
__global__ __launch_bounds__(256) void decode_kernel(const float* __restrict__ cval,
                                                     const int* __restrict__ cidx,
                                                     const float* __restrict__ WdT,
                                                     const float* __restrict__ bias,
                                                     float* __restrict__ xhat) {
  __shared__ float sv[TOPK];
  __shared__ int   si[TOPK];
  const int row = blockIdx.x;
  const int tid = threadIdx.x;
  if (tid < TOPK) {
    sv[tid] = cval[(size_t)row * TOPK + tid];
    si[tid] = cidx[(size_t)row * TOPK + tid];
  }
  __syncthreads();

  float acc[8];
  #pragma unroll
  for (int i = 0; i < 8; ++i) acc[i] = bias[tid + 256 * i];

  for (int j = 0; j < TOPK; ++j) {
    const float v = sv[j];
    if (v != 0.0f) {
      const float* wr = WdT + (size_t)si[j] * D_IN;
      #pragma unroll
      for (int i = 0; i < 8; ++i) acc[i] += v * wr[tid + 256 * i];
    }
  }
  #pragma unroll
  for (int i = 0; i < 8; ++i)
    xhat[(size_t)row * D_IN + tid + 256 * i] = acc[i];
}

// ---------------------------------------------------------------------------
// Launch
// ---------------------------------------------------------------------------
extern "C" void kernel_launch(void* const* d_in, const int* in_sizes, int n_in,
                              void* d_out, int out_size, void* d_ws, size_t ws_size,
                              hipStream_t stream) {
  const float* x     = (const float*)d_in[0];
  const float* bias  = (const float*)d_in[1];
  const float* W_enc = (const float*)d_in[2];
  const float* b_enc = (const float*)d_in[3];
  const float* W_dec = (const float*)d_in[4];
  // d_in[5] is k == 64, fixed at compile time.

  char*   ws   = (char*)d_ws;
  __bf16* xb   = (__bf16*)(ws + OFF_XB);
  __bf16* wb   = (__bf16*)(ws + OFF_WB);
  float*  pre  = (float*) (ws + OFF_PRE);
  float*  WdT  = (float*) (ws + OFF_WDT);
  float*  cval = (float*) (ws + OFF_CV);
  int*    cidx = (int*)   (ws + OFF_CI);

  float* xhat = (float*)d_out;                       // [8192 x 2048]
  float* f    = xhat + (size_t)NTOK * D_IN;          // [8192 x 16384]

  prep_x_kernel<<<2048, 256, 0, stream>>>(x, bias, xb);
  prep_w_kernel<<<4096, 256, 0, stream>>>(W_enc, wb);
  transpose_wdec_kernel<<<dim3(D_HID / 32, D_IN / 32), 256, 0, stream>>>(W_dec, WdT);
  enc_gemm_kernel<<<dim3(D_HID / 128, NTOK / 128), 256, 0, stream>>>(xb, wb, b_enc, pre);
  topk_kernel<<<NTOK, 256, 0, stream>>>(pre, f, cval, cidx);
  decode_kernel<<<NTOK, 256, 0, stream>>>(cval, cidx, WdT, bias, xhat);
}
